// SATO_block_36103495090767
// MI455X (gfx1250) — compile-verified, matrix operand order checked by reading, not run on hardware
//
#include <hip/hip_runtime.h>
#include <hip/hip_bf16.h>
#include <math.h>

// ---------------- problem constants ----------------
#define Bc   8
#define Nn   16384
#define Cc   256
#define Hh   8
#define DHd  32
#define Gg   32
#define Pp   32
#define NPQ  516            // patch rows
#define NPAD (NPQ * Pp)     // 16512 padded tokens
#define HIDc 1024
#define SCALE_ATT 0.17677669529663687f   // 1/sqrt(32)

typedef __attribute__((ext_vector_type(16))) __bf16 v16bf;
typedef __attribute__((ext_vector_type(8)))  float v8f;

__device__ __forceinline__ unsigned short f32_to_bf16(float f) {
  unsigned u = __float_as_uint(f);
  u += 0x7FFFu + ((u >> 16) & 1u);
  return (unsigned short)(u >> 16);
}
__device__ __forceinline__ float bf16_to_f32(unsigned short h) {
  return __uint_as_float(((unsigned)h) << 16);
}
__device__ __forceinline__ unsigned pack_bf16x2(float a, float b) {
  return (unsigned)f32_to_bf16(a) | ((unsigned)f32_to_bf16(b) << 16);
}
__device__ __forceinline__ float wave_sum32(float x) {
  for (int o = 16; o > 0; o >>= 1) x += __shfl_xor(x, o, 32);
  return x;
}

// ---- CDNA5 async global->LDS DMA (ASYNCcnt-tracked) ----
// Low 32 bits of a flat shared-aperture pointer are the LDS byte offset
// (see ISA section 10.2).
__device__ __forceinline__ void async_g2l_b128(void* lds_ptr, const void* gptr) {
  unsigned lds_off = (unsigned)(size_t)lds_ptr;
  asm volatile("global_load_async_to_lds_b128 %0, %1, off"
               :: "v"(lds_off), "v"(gptr) : "memory");
}
__device__ __forceinline__ void wait_async0() {
  asm volatile("s_wait_asynccnt 0" ::: "memory");
}

// ================= weight pack: fp32 [K,N] -> bf16 [N,K] =================
__global__ void pack_wT(const float* __restrict__ W, unsigned short* __restrict__ Wt,
                        int K, int Nout) {
  size_t i = (size_t)blockIdx.x * 256 + threadIdx.x;
  if (i >= (size_t)K * Nout) return;
  int k = (int)(i / Nout), n = (int)(i % Nout);
  Wt[(size_t)n * K + k] = f32_to_bf16(W[i]);
}

__global__ void zero_f32(float* p, int n) {
  int i = blockIdx.x * 256 + threadIdx.x;
  if (i < n) p[i] = 0.f;
}

// ================= generic bf16 WMMA GEMM (double-buffered async LDS) ========
// A[M,K] bf16 row-major, Wt[N,K] bf16 (transposed weight).
// epilogue: +bias, optional exact-GELU, optional fp32 residual add,
// stores fp32 (outF) and/or bf16 (outB).
__global__ __launch_bounds__(256) void gemm_bf16(
    const unsigned short* __restrict__ A, const unsigned short* __restrict__ Wt,
    const float* __restrict__ bias, float* __restrict__ outF,
    unsigned short* __restrict__ outB, const float* __restrict__ resid,
    int M, int K, int Nout, int act) {
  __shared__ unsigned short As[2][128 * 32];
  __shared__ unsigned short Bs[2][64 * 32];
  const int row0 = blockIdx.y * 128, col0 = blockIdx.x * 64;
  const int t = threadIdx.x, lane = t & 31, wave = t >> 5;
  const int wm = wave & 1, wn = wave >> 1;
  const int sr = t >> 1, scp = (t & 1) * 16;

  // stage one 32-deep K panel into buffer `buf` via async DMA (no VGPR staging)
  auto stage = [&](int buf, int k0) {
    const unsigned short* ga = A + (size_t)(row0 + sr) * K + k0 + scp;
    async_g2l_b128(&As[buf][sr * 32 + scp], ga);
    async_g2l_b128(&As[buf][sr * 32 + scp + 8], ga + 8);
    if (t < 128) {
      const unsigned short* gb = Wt + (size_t)(col0 + sr) * K + k0 + scp;
      async_g2l_b128(&Bs[buf][sr * 32 + scp], gb);
      async_g2l_b128(&Bs[buf][sr * 32 + scp + 8], gb + 8);
    }
  };

  v8f acc[4] = {};
  const int nk = K / 32;
  stage(0, 0);
  wait_async0();
  __syncthreads();
  const int am = lane & 15, ak = (lane >> 4) * 16;
  for (int kk = 0; kk < nk; ++kk) {
    const int buf = kk & 1;
    if (kk + 1 < nk) stage(buf ^ 1, (kk + 1) * 32);   // DMA overlaps WMMA below
    v16bf bfr = *(const v16bf*)(&Bs[buf][(wn * 16 + am) * 32 + ak]);
#pragma unroll
    for (int i = 0; i < 4; ++i) {
      v16bf afr = *(const v16bf*)(&As[buf][(wm * 64 + i * 16 + am) * 32 + ak]);
      acc[i] = __builtin_amdgcn_wmma_f32_16x16x32_bf16(false, afr, false, bfr,
                                                       (short)0, acc[i], false, false);
    }
    wait_async0();     // next panel landed in LDS
    __syncthreads();   // all waves done reading `buf` before it is re-staged
  }
  const int nc = col0 + wn * 16 + (lane & 15);
  const float bv = bias ? bias[nc] : 0.f;
#pragma unroll
  for (int i = 0; i < 4; ++i) {
    int mbase = row0 + wm * 64 + i * 16 + 8 * (lane >> 4);
#pragma unroll
    for (int v = 0; v < 8; ++v) {
      int mr = mbase + v;
      float x = acc[i][v] + bv;
      if (act == 1) x = 0.5f * x * (1.f + erff(x * 0.70710678118654752f));
      size_t off = (size_t)mr * Nout + nc;
      if (resid) x += resid[off];
      if (outF) outF[off] = x;
      if (outB) outB[off] = f32_to_bf16(x);
    }
  }
}

// ================= LN1: fx -> fxn (f32 + bf16) =================
__global__ __launch_bounds__(256) void ln1_kernel(const float* __restrict__ fx,
    const float* __restrict__ g, const float* __restrict__ bta,
    float* __restrict__ fxn, unsigned short* __restrict__ fxnb) {
  const int row = blockIdx.x * 8 + (threadIdx.x >> 5);
  const int lane = threadIdx.x & 31;
  const float* src = fx + (size_t)row * Cc + lane * 8;
  float x[8];
  float4 v0 = *(const float4*)src, v1 = *(const float4*)(src + 4);
  x[0]=v0.x; x[1]=v0.y; x[2]=v0.z; x[3]=v0.w; x[4]=v1.x; x[5]=v1.y; x[6]=v1.z; x[7]=v1.w;
  float s = 0.f, s2 = 0.f;
#pragma unroll
  for (int j = 0; j < 8; ++j) { s += x[j]; s2 += x[j] * x[j]; }
  s = wave_sum32(s); s2 = wave_sum32(s2);
  float m = s * (1.f / Cc), var = s2 * (1.f / Cc) - m * m;
  float rs = rsqrtf(var + 1e-5f);
  float y[8];
#pragma unroll
  for (int j = 0; j < 8; ++j) y[j] = (x[j] - m) * rs * g[lane * 8 + j] + bta[lane * 8 + j];
  float* dst = fxn + (size_t)row * Cc + lane * 8;
  *(float4*)dst = make_float4(y[0], y[1], y[2], y[3]);
  *(float4*)(dst + 4) = make_float4(y[4], y[5], y[6], y[7]);
  uint4 pb = { pack_bf16x2(y[0], y[1]), pack_bf16x2(y[2], y[3]),
               pack_bf16x2(y[4], y[5]), pack_bf16x2(y[6], y[7]) };
  *(uint4*)(fxnb + (size_t)row * Cc + lane * 8) = pb;
}

// ================= slice softmax: sw[b,h,n,g] =================
__global__ __launch_bounds__(256) void slice_softmax(const float* __restrict__ xmid,
    const float* __restrict__ Wslice, const float* __restrict__ bslice,
    const float* __restrict__ temp, float* __restrict__ sw) {
  __shared__ float Ws[32 * 32];
  __shared__ float bs[32];
  int t = threadIdx.x;
  for (int i = t; i < 1024; i += 256) Ws[i] = Wslice[i];
  if (t < 32) bs[t] = bslice[t];
  __syncthreads();
  size_t gid = (size_t)blockIdx.x * 256 + t;       // over B*H*N
  int n = (int)(gid % Nn);
  size_t bh = gid / Nn;
  int h = (int)(bh % Hh), b = (int)(bh / Hh);
  const float* xr = xmid + ((size_t)b * Nn + n) * Cc + h * 32;
  float xv[32];
#pragma unroll
  for (int e = 0; e < 32; ++e) xv[e] = xr[e];
  float lg[32];
#pragma unroll
  for (int gI = 0; gI < 32; ++gI) lg[gI] = bs[gI];
  for (int e = 0; e < 32; ++e) {
    float xe = xv[e];
#pragma unroll
    for (int gI = 0; gI < 32; ++gI) lg[gI] += xe * Ws[e * 32 + gI];
  }
  float tinv = 1.f / temp[h], mx = -1e30f;
#pragma unroll
  for (int gI = 0; gI < 32; ++gI) { lg[gI] *= tinv; mx = fmaxf(mx, lg[gI]); }
  float den = 0.f;
#pragma unroll
  for (int gI = 0; gI < 32; ++gI) { lg[gI] = expf(lg[gI] - mx); den += lg[gI]; }
  float inv = 1.f / den;
  float* dst = sw + (bh * Nn + n) * Gg;
#pragma unroll
  for (int gI = 0; gI < 32; ++gI) dst[gI] = lg[gI] * inv;
}

// ================= st / snorm accumulation =================
__global__ __launch_bounds__(256) void st_accum(const float* __restrict__ fxmid,
    const float* __restrict__ sw, float* __restrict__ st, float* __restrict__ snorm) {
  __shared__ float fs[128 * 32];
  __shared__ float ss[128 * 32];
  const int bh = blockIdx.x, b = bh >> 3, h = bh & 7;
  const int n0 = blockIdx.y * 128, t = threadIdx.x;
  for (int i = t; i < 4096; i += 256) {
    int n = i >> 5, d = i & 31;
    fs[i] = fxmid[((size_t)b * Nn + n0 + n) * Cc + h * 32 + d];
    ss[i] = sw[(((size_t)b * Hh + h) * Nn + n0 + n) * Gg + d];
  }
  __syncthreads();
  const int gI = t >> 3, dbase = (t & 7) * 4;
  float a0 = 0, a1 = 0, a2 = 0, a3 = 0, sn = 0;
  for (int n = 0; n < 128; ++n) {
    float p = ss[n * 32 + gI];
    const float* f = fs + n * 32 + dbase;
    a0 += p * f[0]; a1 += p * f[1]; a2 += p * f[2]; a3 += p * f[3];
    sn += p;
  }
  float* so = st + ((size_t)bh * 32 + gI) * 32 + dbase;
  atomicAdd(so + 0, a0); atomicAdd(so + 1, a1);
  atomicAdd(so + 2, a2); atomicAdd(so + 3, a3);
  if (dbase == 0) atomicAdd(&snorm[bh * 32 + gI], sn);
}

// ================= tiny per-(b,h) slice attention (G=32) =================
__global__ __launch_bounds__(32) void tiny_attn(const float* __restrict__ st,
    const float* __restrict__ snorm, const float* __restrict__ Wq,
    const float* __restrict__ Wk, const float* __restrict__ Wv,
    float* __restrict__ ost) {
  __shared__ float sts[32 * 32], qs[32 * 32], ks[32 * 32], vs[32 * 32];
  const int bh = blockIdx.x, gI = threadIdx.x;
  float ninv = 1.f / (snorm[bh * 32 + gI] + 1e-5f);
  for (int d = 0; d < 32; ++d) sts[gI * 32 + d] = st[((size_t)bh * 32 + gI) * 32 + d] * ninv;
  __syncthreads();
  for (int d = 0; d < 32; ++d) {
    float aq = 0, akk = 0, av = 0;
    for (int e = 0; e < 32; ++e) {
      float s = sts[gI * 32 + e];
      aq += s * Wq[e * 32 + d]; akk += s * Wk[e * 32 + d]; av += s * Wv[e * 32 + d];
    }
    qs[gI * 32 + d] = aq; ks[gI * 32 + d] = akk; vs[gI * 32 + d] = av;
  }
  __syncthreads();
  float sc[32], mx = -1e30f;
  for (int j = 0; j < 32; ++j) {
    float a = 0;
    for (int d = 0; d < 32; ++d) a += qs[gI * 32 + d] * ks[j * 32 + d];
    sc[j] = a * SCALE_ATT; mx = fmaxf(mx, sc[j]);
  }
  float den = 0;
  for (int j = 0; j < 32; ++j) { sc[j] = expf(sc[j] - mx); den += sc[j]; }
  float inv = 1.f / den;
  float o[32] = {};
  for (int j = 0; j < 32; ++j) {
    float w = sc[j] * inv;
    for (int d = 0; d < 32; ++d) o[d] += w * vs[j * 32 + d];
  }
  for (int d = 0; d < 32; ++d) ost[((size_t)bh * 32 + gI) * 32 + d] = o[d];
}

// ================= out_x = sw @ ost  (bf16 out, [B,N,C]) =================
__global__ __launch_bounds__(256) void out_x_kernel(const float* __restrict__ sw,
    const float* __restrict__ ost, unsigned short* __restrict__ outx) {
  __shared__ float os[32 * 32];
  const int bh = blockIdx.x, b = bh >> 3, h = bh & 7;
  const int n = blockIdx.y * 256 + threadIdx.x;
  for (int i = threadIdx.x; i < 1024; i += 256) os[i] = ost[(size_t)bh * 1024 + i];
  __syncthreads();
  const float* swr = sw + (((size_t)b * Hh + h) * Nn + n) * Gg;
  float acc[32] = {};
  for (int gI = 0; gI < 32; ++gI) {
    float p = swr[gI];
#pragma unroll
    for (int d = 0; d < 32; ++d) acc[d] += p * os[gI * 32 + d];
  }
  uint4* dst = (uint4*)(outx + ((size_t)b * Nn + n) * Cc + h * 32);
#pragma unroll
  for (int q = 0; q < 4; ++q)
    dst[q] = make_uint4(pack_bf16x2(acc[q*8+0], acc[q*8+1]), pack_bf16x2(acc[q*8+2], acc[q*8+3]),
                        pack_bf16x2(acc[q*8+4], acc[q*8+5]), pack_bf16x2(acc[q*8+6], acc[q*8+7]));
}

// ================= build xp: LN1'(fxn-fx1)[order] into patch layout =================
__global__ __launch_bounds__(256) void build_xp(const float* __restrict__ fxn,
    const float* __restrict__ fx1, const int* __restrict__ order,
    const float* __restrict__ g, const float* __restrict__ bta,
    unsigned short* __restrict__ xp) {
  const int b = blockIdx.y;
  const int jf = blockIdx.x * 8 + (threadIdx.x >> 5);   // 0..16511
  const int lane = threadIdx.x & 31;
  const int r = jf >> 5, c = jf & 31;
  const int m = (r >> 2) * 128 + (r & 3) + 4 * c;
  unsigned short* dst = xp + ((size_t)b * NPAD + jf) * Cc + lane * 8;
  if (m >= Nn) { *(uint4*)dst = make_uint4(0, 0, 0, 0); return; }
  const int src = order[(size_t)b * Nn + m];
  const float* pa = fxn + ((size_t)b * Nn + src) * Cc + lane * 8;
  const float* pb = fx1 + ((size_t)b * Nn + src) * Cc + lane * 8;
  float x[8];
#pragma unroll
  for (int j = 0; j < 8; ++j) x[j] = pa[j] - pb[j];
  float s = 0.f, s2 = 0.f;
#pragma unroll
  for (int j = 0; j < 8; ++j) { s += x[j]; s2 += x[j] * x[j]; }
  s = wave_sum32(s); s2 = wave_sum32(s2);
  float mm = s * (1.f / Cc), rs = rsqrtf(s2 * (1.f / Cc) - mm * mm + 1e-5f);
  float y[8];
#pragma unroll
  for (int j = 0; j < 8; ++j) y[j] = (x[j] - mm) * rs * g[lane * 8 + j] + bta[lane * 8 + j];
  *(uint4*)dst = make_uint4(pack_bf16x2(y[0], y[1]), pack_bf16x2(y[2], y[3]),
                            pack_bf16x2(y[4], y[5]), pack_bf16x2(y[6], y[7]));
}

// ================= patch attention (WMMA), writes o2 already inverse-permuted ====
__global__ __launch_bounds__(256) void patch_attn(const unsigned short* __restrict__ qkv,
                                                  unsigned short* __restrict__ o2p) {
  __shared__ unsigned short sh[8 * 3 * 1024];  // per-wave: S(1K) | P(1K) | vT(1K)
  const int b = blockIdx.y, r = blockIdx.x;    // r < 512: all tokens real
  const int lane = threadIdx.x & 31, h = threadIdx.x >> 5;
  unsigned short* Sm = sh + h * 3 * 1024;
  unsigned short* Pm = Sm + 1024;
  unsigned short* Vt = Sm + 2048;
  const size_t rowbase = ((size_t)b * NPQ + r) * Pp;
  // stage v transposed
  {
    const unsigned short* vp = qkv + (rowbase + lane) * 768 + 512 + h * 32;
#pragma unroll
    for (int d = 0; d < 32; ++d) Vt[d * 32 + lane] = vp[d];
  }
  // S = q k^T
  const int am = lane & 15, ak = (lane >> 4) * 16;
  v16bf afr[2], bfr[2];
#pragma unroll
  for (int mi = 0; mi < 2; ++mi)
    afr[mi] = *(const v16bf*)(qkv + (rowbase + mi * 16 + am) * 768 + 0 + h * 32 + ak);
#pragma unroll
  for (int ni = 0; ni < 2; ++ni)
    bfr[ni] = *(const v16bf*)(qkv + (rowbase + ni * 16 + am) * 768 + 256 + h * 32 + ak);
  v8f sacc[2][2] = {};
#pragma unroll
  for (int mi = 0; mi < 2; ++mi)
#pragma unroll
    for (int ni = 0; ni < 2; ++ni)
      sacc[mi][ni] = __builtin_amdgcn_wmma_f32_16x16x32_bf16(false, afr[mi], false, bfr[ni],
                                                             (short)0, sacc[mi][ni], false, false);
#pragma unroll
  for (int mi = 0; mi < 2; ++mi)
#pragma unroll
    for (int ni = 0; ni < 2; ++ni)
#pragma unroll
      for (int v = 0; v < 8; ++v) {
        int pr = mi * 16 + v + 8 * (lane >> 4), pc = ni * 16 + am;
        Sm[pr * 32 + pc] = f32_to_bf16(sacc[mi][ni][v] * SCALE_ATT);
      }
  __syncthreads();
  // softmax rows (lane = row)
  {
    float sc[32], mx = -1e30f;
#pragma unroll
    for (int j = 0; j < 32; ++j) { sc[j] = bf16_to_f32(Sm[lane * 32 + j]); mx = fmaxf(mx, sc[j]); }
    float den = 0.f;
#pragma unroll
    for (int j = 0; j < 32; ++j) { sc[j] = expf(sc[j] - mx); den += sc[j]; }
    float inv = 1.f / den;
#pragma unroll
    for (int j = 0; j < 32; ++j) Pm[lane * 32 + j] = f32_to_bf16(sc[j] * inv);
  }
  __syncthreads();
  // O = P @ v
  v16bf pfr[2], vfr[2];
#pragma unroll
  for (int mi = 0; mi < 2; ++mi) pfr[mi] = *(const v16bf*)(Pm + (mi * 16 + am) * 32 + ak);
#pragma unroll
  for (int ni = 0; ni < 2; ++ni) vfr[ni] = *(const v16bf*)(Vt + (ni * 16 + am) * 32 + ak);
  v8f oacc[2][2] = {};
#pragma unroll
  for (int mi = 0; mi < 2; ++mi)
#pragma unroll
    for (int ni = 0; ni < 2; ++ni)
      oacc[mi][ni] = __builtin_amdgcn_wmma_f32_16x16x32_bf16(false, pfr[mi], false, vfr[ni],
                                                             (short)0, oacc[mi][ni], false, false);
  __syncthreads();
#pragma unroll
  for (int mi = 0; mi < 2; ++mi)
#pragma unroll
    for (int ni = 0; ni < 2; ++ni)
#pragma unroll
      for (int v = 0; v < 8; ++v) {
        int pr = mi * 16 + v + 8 * (lane >> 4), pc = ni * 16 + am;
        Sm[pr * 32 + pc] = f32_to_bf16(oacc[mi][ni][v]);   // reuse Sm as O
      }
  __syncthreads();
  // scatter row p=lane to original token m (inverse permutation, closed form)
  {
    const int m = (r >> 2) * 128 + (r & 3) + 4 * lane;     // always < N for r<512
    uint4* dst = (uint4*)(o2p + ((size_t)b * Nn + m) * Cc + h * 32);
    const uint4* s = (const uint4*)(Sm + lane * 32);
    dst[0] = s[0]; dst[1] = s[1]; dst[2] = s[2]; dst[3] = s[3];
  }
}

// ================= h = LN3(x0 + fx1 + gate*LN2(yout)[inverse]) -> bf16 ========
__global__ __launch_bounds__(256) void h_kernel(const float* __restrict__ yout,
    const float* __restrict__ fx, const float* __restrict__ fx1,
    const int* __restrict__ inverse, const float* __restrict__ g2,
    const float* __restrict__ b2v, const float* __restrict__ g3,
    const float* __restrict__ b3v, const float* __restrict__ gate,
    unsigned short* __restrict__ hout) {
  const int b = blockIdx.y;
  const int i = blockIdx.x * 8 + (threadIdx.x >> 5);
  const int lane = threadIdx.x & 31;
  const int src = inverse[(size_t)b * Nn + i];
  const float* yr = yout + ((size_t)b * Nn + src) * Cc + lane * 8;
  float x[8];
#pragma unroll
  for (int j = 0; j < 8; ++j) x[j] = yr[j];
  float s = 0.f, s2 = 0.f;
#pragma unroll
  for (int j = 0; j < 8; ++j) { s += x[j]; s2 += x[j] * x[j]; }
  s = wave_sum32(s); s2 = wave_sum32(s2);
  float m = s * (1.f / Cc), rs = rsqrtf(s2 * (1.f / Cc) - m * m + 1e-5f);
  const float gt = gate[0];
  const float* fr = fx + ((size_t)b * Nn + i) * Cc + lane * 8;
  const float* f1 = fx1 + ((size_t)b * Nn + i) * Cc + lane * 8;
  float tv[8];
#pragma unroll
  for (int j = 0; j < 8; ++j) {
    float f2 = (x[j] - m) * rs * g2[lane * 8 + j] + b2v[lane * 8 + j];
    tv[j] = fr[j] + f1[j] + gt * f2;
  }
  s = 0.f; s2 = 0.f;
#pragma unroll
  for (int j = 0; j < 8; ++j) { s += tv[j]; s2 += tv[j] * tv[j]; }
  s = wave_sum32(s); s2 = wave_sum32(s2);
  m = s * (1.f / Cc); rs = rsqrtf(s2 * (1.f / Cc) - m * m + 1e-5f);
  float y[8];
#pragma unroll
  for (int j = 0; j < 8; ++j) y[j] = (tv[j] - m) * rs * g3[lane * 8 + j] + b3v[lane * 8 + j];
  *(uint4*)(hout + ((size_t)b * Nn + i) * Cc + lane * 8) =
      make_uint4(pack_bf16x2(y[0], y[1]), pack_bf16x2(y[2], y[3]),
                 pack_bf16x2(y[4], y[5]), pack_bf16x2(y[6], y[7]));
}

// =====================================================================
extern "C" void kernel_launch(void* const* d_in, const int* in_sizes, int n_in,
                              void* d_out, int out_size, void* d_ws, size_t ws_size,
                              hipStream_t stream) {
  const float* fx      = (const float*)d_in[0];
  const int*   order   = (const int*)d_in[1];
  const int*   inverse = (const int*)d_in[2];
  const float* ln1_g = (const float*)d_in[3],  *ln1_b = (const float*)d_in[4];
  const float* pa_Wfx = (const float*)d_in[5], *pa_bfx = (const float*)d_in[6];
  const float* pa_Wx = (const float*)d_in[7],  *pa_bx = (const float*)d_in[8];
  const float* pa_Wslice = (const float*)d_in[9], *pa_bslice = (const float*)d_in[10];
  const float* pa_temp = (const float*)d_in[11];
  const float* pa_Wq = (const float*)d_in[12], *pa_Wk = (const float*)d_in[13], *pa_Wv = (const float*)d_in[14];
  const float* pa_Wout = (const float*)d_in[15], *pa_bout = (const float*)d_in[16];
  const float* lln1_g = (const float*)d_in[17], *lln1_b = (const float*)d_in[18];
  const float* lln2_g = (const float*)d_in[19], *lln2_b = (const float*)d_in[20];
  const float* sa_Wqkv = (const float*)d_in[21];
  const float* sa_Wout = (const float*)d_in[22], *sa_bout = (const float*)d_in[23];
  const float* ln3_g = (const float*)d_in[24], *ln3_b = (const float*)d_in[25];
  const float* mlp_W1 = (const float*)d_in[26], *mlp_b1 = (const float*)d_in[27];
  const float* mlp_W2 = (const float*)d_in[28], *mlp_b2 = (const float*)d_in[29];
  const float* gate = (const float*)d_in[30];
  float* out = (float*)d_out;

  char* base = (char*)d_ws;
  size_t off = 0;
  auto alloc = [&](size_t bytes) { size_t o = off; off += (bytes + 255) & ~(size_t)255; return o; };

  const size_t MN = (size_t)Bc * Nn;            // 131072 token rows
  const size_t MP = (size_t)Bc * NPAD;          // 132096 padded rows

  unsigned short* WfxT  = (unsigned short*)(base + alloc((size_t)Cc * Cc * 2));
  unsigned short* WxT   = (unsigned short*)(base + alloc((size_t)Cc * Cc * 2));
  unsigned short* WoutT = (unsigned short*)(base + alloc((size_t)Cc * Cc * 2));
  unsigned short* WqkvT = (unsigned short*)(base + alloc((size_t)Cc * 3 * Cc * 2));
  unsigned short* WsoT  = (unsigned short*)(base + alloc((size_t)Cc * Cc * 2));
  unsigned short* W1T   = (unsigned short*)(base + alloc((size_t)Cc * HIDc * 2));
  unsigned short* W2T   = (unsigned short*)(base + alloc((size_t)HIDc * Cc * 2));
  float*          fxn   = (float*)(base + alloc(MN * Cc * 4));
  unsigned short* fxnb  = (unsigned short*)(base + alloc(MN * Cc * 2));
  float*          bufC  = (float*)(base + alloc(MN * Cc * 4));   // x_mid -> yout
  float*          bufD  = (float*)(base + alloc(MN * Cc * 4));   // fx_mid -> fx1
  float*          sw    = (float*)(base + alloc((size_t)Bc * Hh * Nn * Gg * 4));
  float*          stb   = (float*)(base + alloc(64 * 1024 * 4));
  float*          snorm = (float*)(base + alloc(64 * 32 * 4));
  float*          ostb  = (float*)(base + alloc(64 * 1024 * 4));
  unsigned short* bufG  = (unsigned short*)(base + alloc(MN * Cc * 2));   // out_x -> o2p
  unsigned short* bufH  = (unsigned short*)(base + alloc(MP * Cc * 2));   // xp -> h
  unsigned short* bufI  = (unsigned short*)(base + alloc(MN * HIDc * 2)); // qkv -> hid
  (void)ws_size; (void)in_sizes; (void)n_in; (void)out_size;

  // 1. pack weights (bf16, transposed)
  auto packs = [&](const float* W, unsigned short* Wt, int K, int No) {
    int n = (int)(((size_t)K * No + 255) / 256);
    pack_wT<<<n, 256, 0, stream>>>(W, Wt, K, No);
  };
  packs(pa_Wfx, WfxT, Cc, Cc);   packs(pa_Wx, WxT, Cc, Cc);
  packs(pa_Wout, WoutT, Cc, Cc); packs(sa_Wqkv, WqkvT, Cc, 3 * Cc);
  packs(sa_Wout, WsoT, Cc, Cc);  packs(mlp_W1, W1T, Cc, HIDc);
  packs(mlp_W2, W2T, HIDc, Cc);

  // 2. LN1
  ln1_kernel<<<MN / 8, 256, 0, stream>>>(fx, ln1_g, ln1_b, fxn, fxnb);

  // 3/4. fx_mid, x_mid
  gemm_bf16<<<dim3(Cc / 64, MN / 128), 256, 0, stream>>>(fxnb, WfxT, pa_bfx, bufD, nullptr, nullptr, (int)MN, Cc, Cc, 0);
  gemm_bf16<<<dim3(Cc / 64, MN / 128), 256, 0, stream>>>(fxnb, WxT, pa_bx, bufC, nullptr, nullptr, (int)MN, Cc, Cc, 0);

  // 5. sw
  slice_softmax<<<(int)((size_t)Bc * Hh * Nn / 256), 256, 0, stream>>>(bufC, pa_Wslice, pa_bslice, pa_temp, sw);

  // 6/7. st, snorm
  zero_f32<<<(64 * 1024 * 2 + 64 * 32 + 255) / 256, 256, 0, stream>>>(stb, 64 * 1024 * 2 + 64 * 32);
  st_accum<<<dim3(Bc * Hh, Nn / 128), 256, 0, stream>>>(bufD, sw, stb, snorm);

  // 8. tiny attention over slices
  tiny_attn<<<Bc * Hh, 32, 0, stream>>>(stb, snorm, pa_Wq, pa_Wk, pa_Wv, ostb);

  // 9. out_x
  out_x_kernel<<<dim3(Bc * Hh, Nn / 256), 256, 0, stream>>>(sw, ostb, bufG);

  // 10. fx1 = out_x @ Wout + b
  gemm_bf16<<<dim3(Cc / 64, MN / 128), 256, 0, stream>>>(bufG, WoutT, pa_bout, bufD, nullptr, nullptr, (int)MN, Cc, Cc, 0);

  // 11. xp = LN(fxn - fx1)[order] in patch layout (+zero pad)
  build_xp<<<dim3(NPAD / 8, Bc), 256, 0, stream>>>(fxn, bufD, order, lln1_g, lln1_b, bufH);

  // 12. qkv
  gemm_bf16<<<dim3(3 * Cc / 64, MP / 128), 256, 0, stream>>>(bufH, WqkvT, nullptr, nullptr, bufI, nullptr, (int)MP, Cc, 3 * Cc, 0);

  // 13. patch attention + inverse patch scatter
  patch_attn<<<dim3(512, Bc), 256, 0, stream>>>(bufI, bufG);

  // 14. y_out
  gemm_bf16<<<dim3(Cc / 64, MN / 128), 256, 0, stream>>>(bufG, WsoT, sa_bout, bufC, nullptr, nullptr, (int)MN, Cc, Cc, 0);

  // 15. h
  h_kernel<<<dim3(Nn / 8, Bc), 256, 0, stream>>>(bufC, fx, bufD, inverse, lln2_g, lln2_b, ln3_g, ln3_b, gate, bufH);

  // 16. MLP up + GELU
  gemm_bf16<<<dim3(HIDc / 64, MN / 128), 256, 0, stream>>>(bufH, W1T, mlp_b1, nullptr, bufI, nullptr, (int)MN, Cc, HIDc, 1);

  // 17. MLP down + residual -> out
  gemm_bf16<<<dim3(Cc / 64, MN / 128), 256, 0, stream>>>(bufI, W2T, mlp_b2, out, nullptr, fx, (int)MN, HIDc, Cc, 0);
}
